// entropy_57346403336670
// MI455X (gfx1250) — matrix operand order repaired and checked by reading, not compiled, not run on hardware
//
#include <hip/hip_runtime.h>
#include <hip/hip_bf16.h>
#include <math.h>

typedef _Float16 f16;
typedef __attribute__((ext_vector_type(16))) _Float16 v16h;
typedef __attribute__((ext_vector_type(8)))  _Float16 v8h;
typedef __attribute__((ext_vector_type(8)))  float    v8f;

#define N_ROWS 16384
#define K_DIM  256
#define EPSN   1e-8f

// ---- order-preserving float<->uint key (for integer atomicMax on floats) ----
__device__ __forceinline__ unsigned f32_key(float f) {
    unsigned u = __float_as_uint(f);
    return (u & 0x80000000u) ? ~u : (u | 0x80000000u);
}
__device__ __forceinline__ float key_f32(unsigned k) {
    unsigned u = (k & 0x80000000u) ? (k & 0x7fffffffu) : ~k;
    return __uint_as_float(u);
}

// ---------------------------------------------------------------------------
// Kernel 1: row-normalize ex/ey to f16, and (re)initialize the max-key arrays
// one block per row (256 threads == K_DIM)
// ---------------------------------------------------------------------------
__global__ void normalize_rows(const float* __restrict__ ex,
                               const float* __restrict__ ey,
                               f16* __restrict__ xn, f16* __restrict__ yn,
                               unsigned* __restrict__ rk, unsigned* __restrict__ ck) {
    const int r = blockIdx.x;
    const int t = threadIdx.x;
    if (t == 0) { rk[r] = 0u; ck[r] = 0u; }   // key 0 == most-negative float

    const float x = ex[(size_t)r * K_DIM + t];
    const float y = ey[(size_t)r * K_DIM + t];
    float sx = x * x, sy = y * y;
#pragma unroll
    for (int m = 16; m; m >>= 1) {
        sx += __shfl_xor(sx, m, 32);
        sy += __shfl_xor(sy, m, 32);
    }
    __shared__ float wx[8], wy[8];
    const int wave = t >> 5, lane = t & 31;
    if (lane == 0) { wx[wave] = sx; wy[wave] = sy; }
    __syncthreads();
    if (wave == 0) {
        float ax = (lane < 8) ? wx[lane] : 0.f;
        float ay = (lane < 8) ? wy[lane] : 0.f;
#pragma unroll
        for (int m = 4; m; m >>= 1) {
            ax += __shfl_xor(ax, m, 32);
            ay += __shfl_xor(ay, m, 32);
        }
        if (lane == 0) { wx[0] = ax; wy[0] = ay; }
    }
    __syncthreads();
    const float nx = fmaxf(sqrtf(wx[0]), EPSN);
    const float ny = fmaxf(sqrtf(wy[0]), EPSN);
    xn[(size_t)r * K_DIM + t] = (f16)(x / nx);
    yn[(size_t)r * K_DIM + t] = (f16)(y / ny);
}

// ---------------------------------------------------------------------------
// Kernel 2: WMMA GEMM C = Xn * Yn^T with fused row-max / col-max reduction.
// Block = 8 waves; each wave computes a 32(M) x 64(N) tile = 2 A-frags x
// 4 B-frags = 8 accumulators. K accumulated over 8 chunks of 32, with
// register double-buffering so next-chunk loads overlap the WMMA burst.
// ---------------------------------------------------------------------------
__global__ __launch_bounds__(256, 1)
void cosmax_wmma(const f16* __restrict__ X, const f16* __restrict__ Y,
                 unsigned* __restrict__ rk, unsigned* __restrict__ ck) {
    __shared__ unsigned sc[64];

    const int t     = threadIdx.x;
    const int wave  = t >> 5;
    const int lane  = t & 31;
    const int half_ = lane >> 4;   // which 16-lane half of the wave
    const int lid   = lane & 15;

    const int i0 = blockIdx.y * 256 + wave * 32;   // M tile base (per wave)
    const int j0 = blockIdx.x * 64;                // N strip base (per block)

    if (t < 64) sc[t] = 0u;
    __syncthreads();

    // A fragment: lane l<16 holds row i0+l, K = 0..15;
    // lanes 16..31 hold the same rows, K = 16..31 (matches B).
    const f16* arow0 = X + (size_t)(i0 +  0 + lid) * K_DIM + half_ * 16;
    const f16* arow1 = X + (size_t)(i0 + 16 + lid) * K_DIM + half_ * 16;
    // B fragment (column n == Y row n): lane l<16 holds K=0..15 of column
    // j0+16t+l; lanes 16..31 hold K=16..31. One contiguous 32B load per lane.
    const f16* brow[4] = {
        Y + (size_t)(j0 +  0 + lid) * K_DIM + half_ * 16,
        Y + (size_t)(j0 + 16 + lid) * K_DIM + half_ * 16,
        Y + (size_t)(j0 + 32 + lid) * K_DIM + half_ * 16,
        Y + (size_t)(j0 + 48 + lid) * K_DIM + half_ * 16,
    };

    auto loadA = [](const f16* p) -> v16h {
        v8h lo = *(const v8h*)(p);
        v8h hi = *(const v8h*)(p + 8);
        v16h a;
#pragma unroll
        for (int i = 0; i < 8; ++i) { a[i] = lo[i]; a[i + 8] = hi[i]; }
        return a;
    };

    v8f acc[2][4] = {};

    // --- prologue: chunk 0 into "current" fragments
    v16h a0 = loadA(arow0);
    v16h a1 = loadA(arow1);
    v16h b[4];
#pragma unroll
    for (int n = 0; n < 4; ++n) b[n] = *(const v16h*)(brow[n]);

#pragma unroll
    for (int kc = 0; kc < K_DIM; kc += 32) {
        // --- prefetch next chunk into the alternate buffer
        v16h na0, na1, nb[4];
        if (kc + 32 < K_DIM) {
            na0 = loadA(arow0 + kc + 32);
            na1 = loadA(arow1 + kc + 32);
#pragma unroll
            for (int n = 0; n < 4; ++n) nb[n] = *(const v16h*)(brow[n] + kc + 32);
        }
        // --- 8 independent WMMAs on the current chunk
#pragma unroll
        for (int n = 0; n < 4; ++n)
            acc[0][n] = __builtin_amdgcn_wmma_f32_16x16x32_f16(
                false, a0, false, b[n], (short)0, acc[0][n], false, false);
#pragma unroll
        for (int n = 0; n < 4; ++n)
            acc[1][n] = __builtin_amdgcn_wmma_f32_16x16x32_f16(
                false, a1, false, b[n], (short)0, acc[1][n], false, false);
        // --- rotate buffers
        a0 = na0; a1 = na1;
#pragma unroll
        for (int n = 0; n < 4; ++n) b[n] = nb[n];
    }

    // ---- row max: C/D layout => VGPR r, lanes 0-15 = row r cols 0-15,
    //                             lanes 16-31 = row r+8 cols 0-15.
#pragma unroll
    for (int m = 0; m < 2; ++m) {
#pragma unroll
        for (int r = 0; r < 8; ++r) {
            float rm = fmaxf(fmaxf(acc[m][0][r], acc[m][1][r]),
                             fmaxf(acc[m][2][r], acc[m][3][r]));
#pragma unroll
            for (int s = 1; s <= 8; s <<= 1) rm = fmaxf(rm, __shfl_xor(rm, s, 32));
            if (lid == 0) atomicMax(&rk[i0 + m * 16 + half_ * 8 + r], f32_key(rm));
        }
    }

    // ---- col max: reduce the 16 accumulator rows (both M tiles) lane-locally,
    // combine the two 16-lane halves, funnel through LDS to cut global atomics.
#pragma unroll
    for (int n = 0; n < 4; ++n) {
        float cm = acc[0][n][0];
#pragma unroll
        for (int r = 1; r < 8; ++r) cm = fmaxf(cm, acc[0][n][r]);
#pragma unroll
        for (int r = 0; r < 8; ++r) cm = fmaxf(cm, acc[1][n][r]);
        cm = fmaxf(cm, __shfl_xor(cm, 16, 32));
        if (half_ == 0) atomicMax(&sc[n * 16 + lid], f32_key(cm));
    }
    __syncthreads();
    if (t < 64) atomicMax(&ck[j0 + t], sc[t]);
}

// ---------------------------------------------------------------------------
// Kernel 3: entropy over row-max (block 0) and col-max (block 1) keys.
// ---------------------------------------------------------------------------
__global__ void entropy_kernel(const unsigned* __restrict__ rk,
                               const unsigned* __restrict__ ck,
                               float* __restrict__ out) {
    const unsigned* keys = (blockIdx.x == 0) ? rk : ck;
    const float inv2s2 = 1.0f / (2.0f * 0.3f * 0.3f);
    const float neg_logz = 0.28503427f;   // -(log(0.3) + 0.5*log(2*pi))

    float s = 0.f;
    for (int i = threadIdx.x; i < N_ROWS; i += blockDim.x) {
        const float v  = key_f32(keys[i]);
        const float lp = -(v - 1.0f) * (v - 1.0f) * inv2s2 + neg_logz;
        s += expf(lp) * lp;
    }
#pragma unroll
    for (int m = 16; m; m >>= 1) s += __shfl_xor(s, m, 32);
    __shared__ float wsum[8];
    const int wave = threadIdx.x >> 5, lane = threadIdx.x & 31;
    if (lane == 0) wsum[wave] = s;
    __syncthreads();
    if (threadIdx.x == 0) {
        float tot = 0.f;
        for (int w = 0; w < 8; ++w) tot += wsum[w];
        out[blockIdx.x] = -tot;
    }
}

// ---------------------------------------------------------------------------
extern "C" void kernel_launch(void* const* d_in, const int* in_sizes, int n_in,
                              void* d_out, int out_size, void* d_ws, size_t ws_size,
                              hipStream_t stream) {
    const float* ex = (const float*)d_in[0];
    const float* ey = (const float*)d_in[1];

    char* ws = (char*)d_ws;
    const size_t mat_bytes = (size_t)N_ROWS * K_DIM * sizeof(f16);  // 8 MB
    f16*      xn = (f16*)ws;
    f16*      yn = (f16*)(ws + mat_bytes);
    unsigned* rk = (unsigned*)(ws + 2 * mat_bytes);
    unsigned* ck = rk + N_ROWS;

    normalize_rows<<<N_ROWS, K_DIM, 0, stream>>>(ex, ey, xn, yn, rk, ck);

    dim3 grid(N_ROWS / 64, N_ROWS / 256);   // (256, 64) blocks
    cosmax_wmma<<<grid, 256, 0, stream>>>(xn, yn, rk, ck);

    entropy_kernel<<<2, 256, 0, stream>>>(rk, ck, (float*)d_out);
}
